// Bi_cross_Attention_90950227460847
// MI455X (gfx1250) — matrix operand order repaired and checked by reading, not compiled, not run on hardware
//
#include <hip/hip_runtime.h>
#include <hip/hip_bf16.h>

#define HH 448
#define WW 448
#define HWSZ (HH * WW)
#define CC 32
#define BB 4
#define NN 20000

typedef __attribute__((ext_vector_type(16))) __bf16 v16bf;
typedef __attribute__((ext_vector_type(8)))  float  v8f;

// SHIFTS rows: (dy, dx)
__device__ __constant__ int SHY[9] = {0, -1, 1, 0, -1, 1, 0, -1, 1};
__device__ __constant__ int SHX[9] = {0,  0, 0, 1,  1, 1, -1, -1, -1};

// ---------------- workspace layout ----------------
constexpr size_t alignup(size_t x) { return (x + 255) & ~(size_t)255; }
constexpr size_t SZ_LUT  = (size_t)2 * BB * HWSZ * sizeof(int);        // 2 sets x B x H*W
constexpr size_t OFF_LUT = 0;
constexpr size_t OFF_NF  = alignup(OFF_LUT + SZ_LUT);                  // bf16 normed feats [2][B][N][32]
constexpr size_t SZ_NF   = (size_t)2 * BB * NN * CC * 2;
constexpr size_t OFF_PROJ = alignup(OFF_NF + SZ_NF);                   // f32 proj [2][B][3][N][32]
constexpr size_t SZ_PROJ  = (size_t)2 * BB * 3 * NN * CC * 4;
constexpr size_t OFF_CW  = alignup(OFF_PROJ + SZ_PROJ);                // bf16 composed W [2][3][32][32]
constexpr size_t SZ_CW   = (size_t)2 * 3 * CC * CC * 2;
constexpr size_t OFF_CB  = alignup(OFF_CW + SZ_CW);                    // f32 composed bias [2][3][32]
constexpr size_t SZ_CB   = (size_t)2 * 3 * CC * 4;
constexpr size_t OFF_POSV = alignup(OFF_CB + SZ_CB);                   // f32 pos@wv^T [2][9][32]

// ---------------- zero the output canvases ----------------
__global__ void zero_kernel(float4* p, long n4) {
    long i = (long)blockIdx.x * blockDim.x + threadIdx.x;
    long stride = (long)gridDim.x * blockDim.x;
    float4 z; z.x = 0.f; z.y = 0.f; z.z = 0.f; z.w = 0.f;
    for (; i < n4; i += stride) p[i] = z;
}

// ---------------- LUT build ----------------
__global__ void lut_init_kernel(int* luts, long n) {
    long i = (long)blockIdx.x * blockDim.x + threadIdx.x;
    long stride = (long)gridDim.x * blockDim.x;
    for (; i < n; i += stride) luts[i] = -1;
}

__global__ void lut_scatter_kernel(int* luts, const int* li_coors, const int* ra_coors) {
    long t = (long)blockIdx.x * blockDim.x + threadIdx.x;
    if (t >= (long)2 * BB * NN) return;
    int set = (int)(t / (BB * NN));
    long rem = t - (long)set * BB * NN;
    int b = (int)(rem / NN);
    int i = (int)(rem % NN);
    const int* c = (set == 0 ? li_coors : ra_coors) + ((size_t)b * NN + i) * 2;
    int y = c[0], x = c[1];
    luts[((size_t)(set * BB + b)) * HWSZ + y * WW + x] = i;
}

// ---------------- layernorm -> bf16 (wave per point, lane = channel) ----------------
__global__ void ln_kernel(const float* li, const float* ra,
                          const float* lw, const float* lb,
                          const float* rw, const float* rb, __bf16* nf) {
    int lane = threadIdx.x & 31;
    int wave = threadIdx.x >> 5;
    long p = (long)blockIdx.x * 8 + wave;           // over 2*B*N points
    if (p >= (long)2 * BB * NN) return;
    int set = (int)(p / ((long)BB * NN));
    long rem = p - (long)set * BB * NN;
    const float* feat = (set ? ra : li) + rem * CC;
    const float* w   = set ? rw : lw;
    const float* bia = set ? rb : lb;
    float xv = feat[lane];
    float s = xv;
    #pragma unroll
    for (int m = 16; m >= 1; m >>= 1) s += __shfl_xor(s, m, 32);
    float mu = s * (1.0f / CC);
    float d = xv - mu;
    float v = d * d;
    #pragma unroll
    for (int m = 16; m >= 1; m >>= 1) v += __shfl_xor(v, m, 32);
    float rinv = rsqrtf(v * (1.0f / CC) + 1e-5f);
    nf[p * CC + lane] = (__bf16)(d * rinv * w[lane] + bia[lane]);
}

// ---------------- compose weights: Wc = w_qkv @ Wmap (32x32, bf16), biases, pos@wv^T ----------------
__global__ void compose_kernel(const float* qkv1_w, const float* qkv1_b,
                               const float* qkv2_w, const float* qkv2_b,
                               const float* pos_w, const float* pos_b,
                               const float* a1_in_w, const float* a1_in_b,
                               const float* a2_in_w, const float* a2_in_b,
                               __bf16* cw, float* cb, float* posv) {
    int t = threadIdx.x;
    if (t >= 64) return;
    int dir = t >> 5, n = t & 31;
    const float* qw = dir ? qkv2_w : qkv1_w;
    const float* qb = dir ? qkv2_b : qkv1_b;
    const float* iw = dir ? a2_in_w : a1_in_w;
    const float* ib = dir ? a2_in_b : a1_in_b;
    for (int mat = 0; mat < 3; ++mat) {
        const float* wrow = iw + (size_t)(mat * CC + n) * CC;   // in_w row (n of q/k/v slice)
        const float* map  = qw + (size_t)mat * CC * CC;         // qkv_w[mat], [j][c]
        const float* mapb = qb + mat * CC;
        for (int c = 0; c < CC; ++c) {
            float s = 0.f;
            for (int j = 0; j < CC; ++j) s += wrow[j] * map[j * CC + c];
            cw[((size_t)(dir * 3 + mat) * CC + n) * CC + c] = (__bf16)s;
        }
        float bs = ib[mat * CC + n];
        for (int j = 0; j < CC; ++j) bs += mapb[j] * wrow[j];
        cb[(dir * 3 + mat) * CC + n] = bs;
        if (mat == 2) {  // v: pos term composed through wv
            for (int s9 = 0; s9 < 9; ++s9) {
                float ps = 0.f;
                for (int j = 0; j < CC; ++j) {
                    float pj = (float)SHY[s9] * pos_w[j * 2 + 0]
                             + (float)SHX[s9] * pos_w[j * 2 + 1] + pos_b[j];
                    ps += pj * wrow[j];
                }
                posv[(dir * 9 + s9) * CC + n] = ps;
            }
        }
    }
}

// ---------------- WMMA projections: q/k/v composite (N x 32 @ 32 x 32, bf16 -> f32) ----------------
// One wave = 16-point tile; 3 matrices x 2 column tiles = 6 x v_wmma_f32_16x16x32_bf16
__global__ void project_kernel(const __bf16* nf, const __bf16* cw, float* proj) {
    int lane = threadIdx.x & 31;
    int wave = threadIdx.x >> 5;
    int tile = blockIdx.x * 8 + wave;
    if (tile >= NN / 16) return;
    int b = blockIdx.y, set = blockIdx.z;

    const __bf16* feat = nf + ((size_t)(set * BB + b) * NN) * CC;
    // A operand: 16x32 bf16 per ISA layout
    int row = tile * 16 + (lane & 15);
    const __bf16* rowp = feat + (size_t)row * CC;
    int kbA = (lane < 16) ? 0 : 8;
    v16bf a;
    #pragma unroll
    for (int i = 0; i < 8; ++i) { a[i] = rowp[kbA + i]; a[8 + i] = rowp[kbA + 16 + i]; }

    int n0 = lane & 15;
    int kbB = (lane < 16) ? 0 : 8;
    float* pbase = proj + ((size_t)(set * BB + b)) * 3 * NN * CC;

    #pragma unroll
    for (int mat = 0; mat < 3; ++mat) {
        // q uses own direction's weights; k/v use the other direction's (this set is their KV)
        int wdir = (mat == 0) ? set : (1 - set);
        const __bf16* W = cw + ((size_t)(wdir * 3 + mat)) * CC * CC;   // [n][k] row-major
        float* out = pbase + (size_t)mat * NN * CC;
        #pragma unroll
        for (int tn = 0; tn < 2; ++tn) {
            int n = n0 + 16 * tn;
            v16bf bm;
            #pragma unroll
            for (int i = 0; i < 8; ++i) { bm[i] = W[n * CC + kbB + i]; bm[8 + i] = W[n * CC + kbB + 16 + i]; }
            v8f c = {};
            c = __builtin_amdgcn_wmma_f32_16x16x32_bf16(false, a, false, bm,
                                                        (short)0, c, false, false);
            int rbase = tile * 16 + 8 * (lane >> 4);
            #pragma unroll
            for (int i = 0; i < 8; ++i) out[(size_t)(rbase + i) * CC + n] = c[i];
        }
    }
}

// ---------------- fused gather + attention + out-proj + scatter (wave per query) ----------------
__global__ void attn_kernel(const float* proj, const int* luts,
                            const int* li_coors, const int* ra_coors,
                            const float* cb, const float* posv,
                            const float* a1_in_b, const float* a2_in_b,
                            const float* a1_out_w, const float* a1_out_b,
                            const float* a2_out_w, const float* a2_out_b,
                            float* out) {
    int lane = threadIdx.x & 31;          // lane = channel; head = lane/16
    int wave = threadIdx.x >> 5;
    int qi = blockIdx.x * 8 + wave;       // 2500 blocks * 8 = N exactly
    int b = blockIdx.y, dir = blockIdx.z;
    int qset = dir, kvset = 1 - dir;

    const int* qcoor = (qset == 0 ? li_coors : ra_coors) + ((size_t)b * NN + qi) * 2;
    const int* lut = luts + ((size_t)(kvset * BB + b)) * HWSZ;
    const float* qproj = proj + ((size_t)(qset  * BB + b)) * 3 * NN * CC;
    const float* kproj = proj + ((size_t)(kvset * BB + b)) * 3 * NN * CC + (size_t)1 * NN * CC;
    const float* vproj = proj + ((size_t)(kvset * BB + b)) * 3 * NN * CC + (size_t)2 * NN * CC;
    const float* cbd  = cb + dir * 3 * CC;
    const float* in_b = dir ? a2_in_b  : a1_in_b;
    const float* ow   = dir ? a2_out_w : a1_out_w;
    const float* ob   = dir ? a2_out_b : a1_out_b;

    float q = qproj[(size_t)qi * CC + lane] + cbd[0 * CC + lane];
    int y = qcoor[0], x = qcoor[1];

    float sc[9], vv[9];
    #pragma unroll
    for (int k = 0; k < 9; ++k) {
        int sy = y + SHY[k], sx = x + SHX[k];
        int gi = -1;
        if (sy >= 0 && sy < HH && sx >= 0 && sx < WW) gi = lut[sy * WW + sx];
        float kk, vk;
        if (gi >= 0) {
            kk = kproj[(size_t)gi * CC + lane] + cbd[1 * CC + lane];
            vk = vproj[(size_t)gi * CC + lane] + cbd[2 * CC + lane]
               + posv[(dir * 9 + k) * CC + lane];
        } else {
            kk = in_b[CC + lane];        // masked -> k row = bk
            vk = in_b[2 * CC + lane];    // masked -> v row = bv
        }
        float p = q * kk;                // per-head dot over 16 lanes
        #pragma unroll
        for (int m = 8; m >= 1; m >>= 1) p += __shfl_xor(p, m, 16);
        sc[k] = p * 0.25f;               // 1/sqrt(HEAD_DIM=16)
        vv[k] = vk;
    }
    float mx = sc[0];
    #pragma unroll
    for (int k = 1; k < 9; ++k) mx = fmaxf(mx, sc[k]);
    float se = 0.f, o = 0.f;
    #pragma unroll
    for (int k = 0; k < 9; ++k) {
        float e = __expf(sc[k] - mx);
        se += e;
        o = fmaf(e, vv[k], o);
    }
    o /= se;

    // out = o @ out_w^T + out_b  (32x32 via cross-lane broadcast)
    float acc = ob[lane];
    #pragma unroll
    for (int c = 0; c < CC; ++c) acc = fmaf(__shfl(o, c, 32), ow[lane * CC + c], acc);

    size_t base = ((size_t)(dir * BB + b)) * CC * HWSZ
                + (size_t)lane * HWSZ + (size_t)y * WW + x;
    out[base] = acc;
}

extern "C" void kernel_launch(void* const* d_in, const int* in_sizes, int n_in,
                              void* d_out, int out_size, void* d_ws, size_t ws_size,
                              hipStream_t stream) {
    const float* li_feats = (const float*)d_in[0];
    const float* ra_feats = (const float*)d_in[1];
    const int*   li_coors = (const int*)d_in[2];
    const int*   ra_coors = (const int*)d_in[3];
    const float* li_nw = (const float*)d_in[4];
    const float* li_nb = (const float*)d_in[5];
    const float* ra_nw = (const float*)d_in[6];
    const float* ra_nb = (const float*)d_in[7];
    const float* qkv1_w = (const float*)d_in[8];
    const float* qkv1_b = (const float*)d_in[9];
    const float* qkv2_w = (const float*)d_in[10];
    const float* qkv2_b = (const float*)d_in[11];
    const float* pos_w  = (const float*)d_in[12];
    const float* pos_b  = (const float*)d_in[13];
    const float* a1_in_w  = (const float*)d_in[14];
    const float* a1_in_b  = (const float*)d_in[15];
    const float* a1_out_w = (const float*)d_in[16];
    const float* a1_out_b = (const float*)d_in[17];
    const float* a2_in_w  = (const float*)d_in[18];
    const float* a2_in_b  = (const float*)d_in[19];
    const float* a2_out_w = (const float*)d_in[20];
    const float* a2_out_b = (const float*)d_in[21];

    char* ws = (char*)d_ws;
    int*    luts = (int*)(ws + OFF_LUT);
    __bf16* nf   = (__bf16*)(ws + OFF_NF);
    float*  proj = (float*)(ws + OFF_PROJ);
    __bf16* cw   = (__bf16*)(ws + OFF_CW);
    float*  cb   = (float*)(ws + OFF_CB);
    float*  posv = (float*)(ws + OFF_POSV);

    // 1) zero the 205 MB of output canvases (bandwidth-dominant step)
    long n4 = (long)out_size / 4;
    zero_kernel<<<4096, 256, 0, stream>>>((float4*)d_out, n4);

    // 2) LUTs
    lut_init_kernel<<<2048, 256, 0, stream>>>(luts, (long)2 * BB * HWSZ);
    lut_scatter_kernel<<<(2 * BB * NN + 255) / 256, 256, 0, stream>>>(luts, li_coors, ra_coors);

    // 3) layernorm -> bf16
    ln_kernel<<<(2 * BB * NN) / 8, 256, 0, stream>>>(li_feats, ra_feats,
                                                     li_nw, li_nb, ra_nw, ra_nb, nf);

    // 4) compose weights (tiny)
    compose_kernel<<<1, 64, 0, stream>>>(qkv1_w, qkv1_b, qkv2_w, qkv2_b, pos_w, pos_b,
                                         a1_in_w, a1_in_b, a2_in_w, a2_in_b, cw, cb, posv);

    // 5) WMMA projections: 1250 tiles of 16 points, 8 waves/block
    project_kernel<<<dim3((NN / 16 + 7) / 8, BB, 2), 256, 0, stream>>>(nf, cw, proj);

    // 6) fused attention + scatter: wave per query point
    attn_kernel<<<dim3(NN / 8, BB, 2), 256, 0, stream>>>(proj, luts, li_coors, ra_coors,
                                                         cb, posv, a1_in_b, a2_in_b,
                                                         a1_out_w, a1_out_b, a2_out_w, a2_out_b,
                                                         (float*)d_out);
}
